// CausalSelfAttention_20615843021476
// MI455X (gfx1250) — compile-verified, matrix operand order checked by reading, not run on hardware
//
#include <hip/hip_runtime.h>
#include <hip/hip_bf16.h>
#include <math.h>

// ---------------------------------------------------------------------------
// CDNA5 (gfx1250) fused causal-window GQA attention block.
// - All matmuls on v_wmma_f32_16x16x32_bf16 (fp32 in, bf16 compute, f32 acc)
// - GEMMs: 2-stage software pipeline (b128 load clusters overlap WMMA)
// - Flash attention: K/V tiles double-buffered in LDS via the Tensor Data
//   Mover (tensor_load_to_lds, TENSORcnt-pipelined) when available.
// ---------------------------------------------------------------------------

typedef __attribute__((ext_vector_type(16))) __bf16       v16bf;
typedef __attribute__((ext_vector_type(8)))  float        v8f;
typedef __attribute__((ext_vector_type(4)))  unsigned int v4u;
typedef __attribute__((ext_vector_type(8)))  int          v8i;
typedef __attribute__((ext_vector_type(4)))  int          v4i;

constexpr int NB    = 2;     // batch
constexpr int NT    = 2048;  // seq len
constexpr int NC    = 1024;  // channels
constexpr int NH    = 16;    // q heads
constexpr int NHKV  = 4;     // kv heads
constexpr int ND    = 64;    // head dim
constexpr int NGATE = 32;    // gate channels

// ISA 7.12.2: 16-bit A 16x32 — lane<16 holds K={0..7,16..23}, lane>=16 holds
// K={8..15,24..31}.  With off = 8*half: element e -> K = off+e (e<8),
// off+16+(e-8) (e>=8): two contiguous 8-float runs => four b128 loads.
__device__ __forceinline__ int kmap(int e, int half) {
  return e + 8 * half + ((e >= 8) ? 8 : 0);
}

struct Raw4 { float4 r0, r1, r2, r3; };

__device__ __forceinline__ Raw4 load_raw(const float* base, int off) {
  Raw4 q;
  const float4* p0 = reinterpret_cast<const float4*>(base + off);
  const float4* p1 = reinterpret_cast<const float4*>(base + off + 16);
  q.r0 = p0[0]; q.r1 = p0[1]; q.r2 = p1[0]; q.r3 = p1[1];
  return q;
}

__device__ __forceinline__ v16bf cvt16(float4 r0, float4 r1, float4 r2, float4 r3) {
  v16bf o;
  o[0]=(__bf16)r0.x; o[1]=(__bf16)r0.y; o[2]=(__bf16)r0.z; o[3]=(__bf16)r0.w;
  o[4]=(__bf16)r1.x; o[5]=(__bf16)r1.y; o[6]=(__bf16)r1.z; o[7]=(__bf16)r1.w;
  o[8]=(__bf16)r2.x; o[9]=(__bf16)r2.y; o[10]=(__bf16)r2.z; o[11]=(__bf16)r2.w;
  o[12]=(__bf16)r3.x; o[13]=(__bf16)r3.y; o[14]=(__bf16)r3.z; o[15]=(__bf16)r3.w;
  return o;
}
__device__ __forceinline__ v16bf cvt16s(float4 r0, float4 r1, float4 r2, float4 r3, float s) {
  v16bf o;
  o[0]=(__bf16)(r0.x*s); o[1]=(__bf16)(r0.y*s); o[2]=(__bf16)(r0.z*s); o[3]=(__bf16)(r0.w*s);
  o[4]=(__bf16)(r1.x*s); o[5]=(__bf16)(r1.y*s); o[6]=(__bf16)(r1.z*s); o[7]=(__bf16)(r1.w*s);
  o[8]=(__bf16)(r2.x*s); o[9]=(__bf16)(r2.y*s); o[10]=(__bf16)(r2.z*s); o[11]=(__bf16)(r2.w*s);
  o[12]=(__bf16)(r3.x*s); o[13]=(__bf16)(r3.y*s); o[14]=(__bf16)(r3.z*s); o[15]=(__bf16)(r3.w*s);
  return o;
}
__device__ __forceinline__ v16bf cvt_raw(const Raw4& q) {
  return cvt16(q.r0, q.r1, q.r2, q.r3);
}
__device__ __forceinline__ v16bf load_frag(const float* base, int off) {
  const Raw4 q = load_raw(base, off);
  return cvt_raw(q);
}

__device__ __forceinline__ float redmax16(float v) {
#pragma unroll
  for (int off = 8; off; off >>= 1) v = fmaxf(v, __shfl_xor(v, off, 32));
  return v;
}
__device__ __forceinline__ float redsum16(float v) {
#pragma unroll
  for (int off = 8; off; off >>= 1) v += __shfl_xor(v, off, 32);
  return v;
}
__device__ __forceinline__ float redsum32(float v) {
#pragma unroll
  for (int off = 16; off; off >>= 1) v += __shfl_xor(v, off, 32);
  return v;
}

// ---------------------------------------------------------------------------
// Tensor Data Mover: stage a 2-D fp32 tile (tile_w x tile_h elements, row
// stride row_stride elems) from global into LDS.  D# per ISA ch.8.
// ---------------------------------------------------------------------------
#if defined(__gfx1250__) && __has_builtin(__builtin_amdgcn_tensor_load_to_lds)
#define USE_TDM 1
__device__ __forceinline__ void tdm_load_tile_2d(unsigned int lds_addr,
                                                 const float* gsrc,
                                                 unsigned int tile_w,
                                                 unsigned int tile_h,
                                                 unsigned int row_stride) {
  const unsigned long long ga = (unsigned long long)(uintptr_t)gsrc;
  v4u g0;
  g0[0] = 1u;                                           // count=1, user mode
  g0[1] = lds_addr;                                     // lds_addr[31:0]
  g0[2] = (unsigned int)(ga & 0xFFFFFFFFu);             // global_addr[31:0]
  g0[3] = (unsigned int)((ga >> 32) & 0x01FFFFFFu)      // global_addr[56:32]
        | 0x80000000u;                                  // type=2 ("image")
  v8i g1;
  g1[0] = (int)(2u << 16);                              // wg_mask=0, data_size=4B
  g1[1] = (int)((tile_w & 0xFFFFu) << 16);              // tensor_dim0[15:0]
  g1[2] = (int)(((unsigned)NT & 0xFFFFu) << 16);        // dim0 hi | tensor_dim1 lo
  g1[3] = (int)(tile_w << 16);                          // dim1 hi | tile_dim0
  g1[4] = (int)(tile_h & 0xFFFFu);                      // tile_dim1 | tile_dim2=0
  g1[5] = (int)row_stride;                              // tensor_dim0_stride lo
  g1[6] = 0;                                            // stride0 hi | stride1 lo
  g1[7] = 0;
  v4i z4 = {0, 0, 0, 0};
#if defined(__clang_major__) && (__clang_major__ >= 23)
  v8i z8 = {0, 0, 0, 0, 0, 0, 0, 0};
  __builtin_amdgcn_tensor_load_to_lds(g0, g1, z4, z4, z8, 0);
#else
  __builtin_amdgcn_tensor_load_to_lds(g0, g1, z4, z4, 0);
#endif
}
#endif

// ---------------------------------------------------------------------------
// C[M,N] = A[M,K] * W[N,K]^T  (fp32 in/out, bf16 WMMA).  One wave computes a
// 16x64 strip; 2-stage pipeline: next k-step's 20 b128 loads are issued
// before the current step's 4 WMMAs.  M%16==0, N%64==0, K%32==0, K>=64.
// ---------------------------------------------------------------------------
__global__ __launch_bounds__(256)
void wmma_gemm_nt(const float* __restrict__ A,
                  const float* __restrict__ W,
                  float* __restrict__ Cout,
                  int M, int N, int K) {
  const int lane   = threadIdx.x & 31;
  const int wave   = threadIdx.x >> 5;
  const int strip  = blockIdx.x * (blockDim.x >> 5) + wave;
  const int nstrip = N >> 6;
  const int mt     = strip / nstrip;
  const int nt     = strip % nstrip;
  if (mt * 16 >= M) return;

  const int m     = lane & 15;
  const int half  = lane >> 4;
  const int base0 = 8 * half;
  const float* arow = A + (size_t)(mt * 16 + m) * K;
  const float* wrow[4];
#pragma unroll
  for (int j = 0; j < 4; ++j)
    wrow[j] = W + (size_t)(nt * 64 + j * 16 + m) * K;

  v8f acc[4] = {};

  // pipeline prologue: stage k-step 0
  Raw4 aC = load_raw(arow, base0);
  Raw4 bC[4];
#pragma unroll
  for (int j = 0; j < 4; ++j) bC[j] = load_raw(wrow[j], base0);

  int k0 = 0;
  for (; k0 < K - 32; k0 += 32) {
    const int off = k0 + 32 + base0;
    __builtin_prefetch(arow + k0 + 160, 0, 1);
    // issue next k-step's loads (independent of current compute)
    Raw4 aN = load_raw(arow, off);
    Raw4 bN[4];
#pragma unroll
    for (int j = 0; j < 4; ++j) bN[j] = load_raw(wrow[j], off);

    // compute current k-step
    const v16bf a = cvt_raw(aC);
#pragma unroll
    for (int j = 0; j < 4; ++j) {
      const v16bf bfr = cvt_raw(bC[j]);
      acc[j] = __builtin_amdgcn_wmma_f32_16x16x32_bf16(false, a, false, bfr,
                                                       (short)0, acc[j], false, false);
    }
    aC = aN;
#pragma unroll
    for (int j = 0; j < 4; ++j) bC[j] = bN[j];
  }
  // epilogue: last k-step
  {
    const v16bf a = cvt_raw(aC);
#pragma unroll
    for (int j = 0; j < 4; ++j) {
      const v16bf bfr = cvt_raw(bC[j]);
      acc[j] = __builtin_amdgcn_wmma_f32_16x16x32_bf16(false, a, false, bfr,
                                                       (short)0, acc[j], false, false);
    }
  }

  // C layout: VGPR r, lane l -> M = r + 8*(l>>4), N = l&15
#pragma unroll
  for (int j = 0; j < 4; ++j)
#pragma unroll
    for (int r = 0; r < 8; ++r)
      Cout[(size_t)(mt * 16 + r + 8 * half) * N + nt * 64 + j * 16 + (lane & 15)] =
          acc[j][r];
}

// ---------------------------------------------------------------------------
// v[b,t,hkv,:] += 2*sigmoid(x[b,t,:32] @ Wgate[hkv,:]) * ve[b,t,hkv,:]
// ---------------------------------------------------------------------------
__global__ void gate_v_kernel(float* __restrict__ v,
                              const float* __restrict__ x,
                              const float* __restrict__ ve,
                              const float* __restrict__ Wg) {
  const int wid  = blockIdx.x * (blockDim.x >> 5) + (threadIdx.x >> 5);
  const int lane = threadIdx.x & 31;
  if (wid >= NB * NT) return;
  const int t = wid % NT, b = wid / NT;

  const float xv = x[((size_t)b * NT + t) * NC + lane];  // lane < 32 == NGATE
  float g[NHKV];
#pragma unroll
  for (int hk = 0; hk < NHKV; ++hk) {
    float p = xv * Wg[hk * NGATE + lane];
    p = redsum32(p);
    g[hk] = 2.0f / (1.0f + __expf(-p));
  }
  float*       vp  = v  + ((size_t)b * NT + t) * NHKV * ND;
  const float* vep = ve + ((size_t)b * NT + t) * NHKV * ND;
  for (int i = lane; i < NHKV * ND; i += 32)
    vp[i] += g[i / ND] * vep[i];
}

// ---------------------------------------------------------------------------
// Rotary + RMSNorm in place; one wave per (b,t,h); lane owns (d, d+32).
// ---------------------------------------------------------------------------
__global__ void rope_rms_kernel(float* __restrict__ buf,
                                const float* __restrict__ cosb,
                                const float* __restrict__ sinb,
                                int nheads, int total) {
  const int wid  = blockIdx.x * (blockDim.x >> 5) + (threadIdx.x >> 5);
  const int lane = threadIdx.x & 31;
  if (wid >= total) return;
  const int h = wid % nheads;
  const int t = (wid / nheads) % NT;
  const int b = wid / (nheads * NT);

  float* p = buf + (((size_t)b * NT + t) * nheads + h) * ND;
  const float c  = cosb[t * 32 + lane];
  const float s  = sinb[t * 32 + lane];
  const float x1 = p[lane], x2 = p[lane + 32];
  const float r1 =  x1 * c + x2 * s;
  const float r2 = -x1 * s + x2 * c;
  const float ss  = redsum32(r1 * r1 + r2 * r2);
  const float inv = rsqrtf(ss * (1.0f / ND) + 1e-6f);
  p[lane]      = r1 * inv;
  p[lane + 32] = r2 * inv;
}

// ---------------------------------------------------------------------------
// Flash attention, sliding causal window.  One single-wave workgroup per
// (b, h, 16-q block).  K/V tiles double-buffered in LDS via TDM: next
// block's tensor loads overlap current block's WMMAs (wait tensorcnt<=2).
// LDS: 2x8KB K + 2x8KB V + 2KB P = 34KB.
// ---------------------------------------------------------------------------
__global__ __launch_bounds__(32)
void flash_attn_kernel(const float* __restrict__ qg,
                       const float* __restrict__ kg,
                       const float* __restrict__ vg,
                       float* __restrict__ yg,
                       const int* __restrict__ win_ptr) {
  __shared__ float ktile[2][32][64];
  __shared__ float vtile[2][32][64];
  __shared__ float ptile[16][32];

  const int lane = threadIdx.x & 31;
  const int task = blockIdx.x;
  const int QB   = NT / 16;
  const int qblk = task % QB;
  const int h    = (task / QB) % NH;
  const int b    = task / (QB * NH);
  if (b >= NB) return;
  const int hkv = h / (NH / NHKV);
  const int win = *win_ptr;

  const int m    = lane & 15;
  const int half = lane >> 4;
  const int nidx = lane & 15;
  const int q0   = qblk * 16;

  const float* kbase = kg + ((size_t)b * NT * NHKV + hkv) * ND;
  const float* vbase = vg + ((size_t)b * NT * NHKV + hkv) * ND;

  // Q fragments (scale 1/sqrt(D)=0.125 folded): two K=32 chunks over D=64.
  v16bf qa[2];
  {
    const float* qrow = qg + (((size_t)b * NT + (q0 + m)) * NH + h) * ND;
#pragma unroll
    for (int c = 0; c < 2; ++c) {
      const Raw4 q = load_raw(qrow, c * 32 + 8 * half);
      qa[c] = cvt16s(q.r0, q.r1, q.r2, q.r3, 0.125f);
    }
  }

  v8f   o[4] = {};
  float row_max[8], row_sum[8];
#pragma unroll
  for (int r = 0; r < 8; ++r) { row_max[r] = -1e30f; row_sum[r] = 0.0f; }

  int ks = q0 + 15 - win; if (ks < 0) ks = 0; ks &= ~31;
  const int kend = q0 + 16;

#if defined(USE_TDM)
  // pipeline prologue: stage first K/V block into buffer 0
  tdm_load_tile_2d((unsigned int)(uintptr_t)&ktile[0][0][0],
                   kbase + (size_t)ks * NHKV * ND, ND, 32, NHKV * ND);
  tdm_load_tile_2d((unsigned int)(uintptr_t)&vtile[0][0][0],
                   vbase + (size_t)ks * NHKV * ND, ND, 32, NHKV * ND);
#endif

  int idx = 0;
  for (int kb0 = ks; kb0 < kend; kb0 += 32, ++idx) {
    const int cur = idx & 1;
    float (*Kt)[64] = ktile[cur];
    float (*Vt)[64] = vtile[cur];

#if defined(USE_TDM)
    if (kb0 + 32 < kend) {
      // WAR: buffer cur^1 was read in the previous block; drain LDS reads first.
      asm volatile("s_wait_dscnt 0" ::: "memory");
      tdm_load_tile_2d((unsigned int)(uintptr_t)&ktile[cur ^ 1][0][0],
                       kbase + (size_t)(kb0 + 32) * NHKV * ND, ND, 32, NHKV * ND);
      tdm_load_tile_2d((unsigned int)(uintptr_t)&vtile[cur ^ 1][0][0],
                       vbase + (size_t)(kb0 + 32) * NHKV * ND, ND, 32, NHKV * ND);
      __builtin_amdgcn_s_wait_tensorcnt((short)2);  // current tiles complete
    } else {
      __builtin_amdgcn_s_wait_tensorcnt((short)0);
    }
    asm volatile("" ::: "memory");
#else
    const float* ksrc = kbase + (size_t)kb0 * NHKV * ND;
    const float* vsrc = vbase + (size_t)kb0 * NHKV * ND;
    for (int i = lane; i < 32 * 16; i += 32) {
      const int row = i >> 4, c4 = i & 15;
      reinterpret_cast<float4*>(&Kt[row][0])[c4] =
          reinterpret_cast<const float4*>(ksrc + (size_t)row * NHKV * ND)[c4];
      reinterpret_cast<float4*>(&Vt[row][0])[c4] =
          reinterpret_cast<const float4*>(vsrc + (size_t)row * NHKV * ND)[c4];
    }
#endif

    // ---- scores: S[16 q x 32 keys] as two 16x16 C tiles ----
    v8f sacc[2] = {};
#pragma unroll
    for (int nh = 0; nh < 2; ++nh) {
      const float* krow = &Kt[nh * 16 + m][0];     // B-frag column n = lane&15
#pragma unroll
      for (int c = 0; c < 2; ++c) {
        const v16bf bfr = load_frag(krow, c * 32 + 8 * half);
        sacc[nh] = __builtin_amdgcn_wmma_f32_16x16x32_bf16(
            false, qa[c], false, bfr, (short)0, sacc[nh], false, false);
      }
    }

    // ---- mask + online softmax (C rows live on lane%16 groups) ----
#pragma unroll
    for (int r = 0; r < 8; ++r) {
      const int i  = q0 + r + 8 * half;
      const int j0 = kb0 + nidx;
      const int j1 = kb0 + 16 + nidx;
      const float s0 = (j0 <= i && j0 >= i - win) ? sacc[0][r] : -1e30f;
      const float s1 = (j1 <= i && j1 >= i - win) ? sacc[1][r] : -1e30f;
      const float bm = redmax16(fmaxf(s0, s1));
      const float nm = fmaxf(row_max[r], bm);
      const float alpha = __expf(row_max[r] - nm);
      const float p0 = (s0 > -5e29f) ? __expf(s0 - nm) : 0.0f;
      const float p1 = (s1 > -5e29f) ? __expf(s1 - nm) : 0.0f;
      row_sum[r] = row_sum[r] * alpha + redsum16(p0 + p1);
      row_max[r] = nm;
#pragma unroll
      for (int t4 = 0; t4 < 4; ++t4) o[t4][r] *= alpha;
      ptile[r + 8 * half][nidx]      = p0;         // C-layout -> LDS
      ptile[r + 8 * half][16 + nidx] = p1;
    }

    // ---- PV: P (16x32) as A fragment, V (32x64) as 4 B fragments ----
    const v16bf pa = load_frag(&ptile[m][0], 8 * half);
#pragma unroll
    for (int t4 = 0; t4 < 4; ++t4) {
      v16bf vf;
#pragma unroll
      for (int e = 0; e < 16; ++e)
        vf[e] = (__bf16)Vt[kmap(e, half)][t4 * 16 + nidx];
      o[t4] = __builtin_amdgcn_wmma_f32_16x16x32_bf16(
          false, pa, false, vf, (short)0, o[t4], false, false);
    }
  }

  // ---- normalize + store y[b, row, h, :] ----
#pragma unroll
  for (int r = 0; r < 8; ++r) {
    const float inv = (row_sum[r] > 0.0f) ? 1.0f / row_sum[r] : 0.0f;
    const int   row = q0 + r + 8 * half;
    float* yrow = yg + (((size_t)b * NT + row) * NH + h) * ND;
#pragma unroll
    for (int t4 = 0; t4 < 4; ++t4)
      yrow[t4 * 16 + nidx] = o[t4][r] * inv;
  }
}

// ---------------------------------------------------------------------------
extern "C" void kernel_launch(void* const* d_in, const int* in_sizes, int n_in,
                              void* d_out, int out_size, void* d_ws, size_t ws_size,
                              hipStream_t stream) {
  (void)in_sizes; (void)n_in; (void)out_size; (void)ws_size;
  const float* x     = (const float*)d_in[0];
  const float* ve    = (const float*)d_in[1];
  const float* cosb  = (const float*)d_in[2];
  const float* sinb  = (const float*)d_in[3];
  const float* Wq    = (const float*)d_in[4];
  const float* Wk    = (const float*)d_in[5];
  const float* Wv    = (const float*)d_in[6];
  const float* Wproj = (const float*)d_in[7];
  const float* Wgate = (const float*)d_in[8];
  const int*   win   = (const int*)d_in[9];
  float* out = (float*)d_out;

  float* qbuf = (float*)d_ws;                               // B*T*H*D
  float* kbuf = qbuf + (size_t)NB * NT * NH   * ND;         // B*T*HKV*D
  float* vbuf = kbuf + (size_t)NB * NT * NHKV * ND;         // B*T*HKV*D
  float* ybuf = vbuf + (size_t)NB * NT * NHKV * ND;         // B*T*C

  const int M = NB * NT;                                    // 4096 rows

  // QKV projections: strips = (M/16)*(N/64), 8 waves per 256-thread block
  wmma_gemm_nt<<<(M / 16) * (NH   * ND / 64) / 8, 256, 0, stream>>>(x, Wq, qbuf, M, NH   * ND, NC);
  wmma_gemm_nt<<<(M / 16) * (NHKV * ND / 64) / 8, 256, 0, stream>>>(x, Wk, kbuf, M, NHKV * ND, NC);
  wmma_gemm_nt<<<(M / 16) * (NHKV * ND / 64) / 8, 256, 0, stream>>>(x, Wv, vbuf, M, NHKV * ND, NC);

  // gate * ve into v
  gate_v_kernel<<<(NB * NT) / 8, 256, 0, stream>>>(vbuf, x, ve, Wgate);

  // rotary + rmsnorm on q and k
  rope_rms_kernel<<<(NB * NT * NH)   / 8, 256, 0, stream>>>(qbuf, cosb, sinb, NH,   NB * NT * NH);
  rope_rms_kernel<<<(NB * NT * NHKV) / 8, 256, 0, stream>>>(kbuf, cosb, sinb, NHKV, NB * NT * NHKV);

  // windowed-causal flash attention (TDM double-buffered K/V, WMMA matmuls)
  flash_attn_kernel<<<NB * NH * (NT / 16), 32, 0, stream>>>(qbuf, kbuf, vbuf, ybuf, win);

  // output projection
  wmma_gemm_nt<<<(M / 16) * (NC / 64) / 8, 256, 0, stream>>>(ybuf, Wproj, out, M, NC, NC);
}
